// DecoderLSTM_9285719294339
// MI455X (gfx1250) — compile-verified, matrix operand order checked by reading, not run on hardware
//
#include <hip/hip_runtime.h>

#define B_DIM 256
#define T_DIM 256
#define I_DIM 1024
#define H_DIM 1024
#define C_DIM 1024
#define F_DIM 512
#define NC_DIM 7
#define G4H   (4 * H_DIM)
#define KC    128   // K-chunk staged in LDS per double-buffer slot

typedef __attribute__((ext_vector_type(16))) __bf16 v16bf;
typedef __attribute__((ext_vector_type(8)))  __bf16 v8bf;
typedef __attribute__((ext_vector_type(8)))  float  v8f;
typedef __attribute__((ext_vector_type(4)))  uint32_t u32x4_t;
typedef __attribute__((ext_vector_type(8)))  int      i32x8_t;
typedef __attribute__((ext_vector_type(4)))  int      i32x4_t;

// ---------------------------------------------------------------------------
// WMMA helper (CDNA5 bf16 16x16x32, f32 accumulate)
// ---------------------------------------------------------------------------
static __device__ inline v8f wmma_bf16(v16bf a, v16bf b, v8f c) {
  return __builtin_amdgcn_wmma_f32_16x16x32_bf16(false, a, false, b, (short)0, c,
                                                 false, false);
}

// Fragment from a row-major bf16 matrix (global). ISA layout:
// lanes 0-15 hold K {0..7, 16..23}+k0, lanes 16-31 hold K {8..15, 24..31}+k0.
static __device__ inline v16bf frag_from_bf16(const __bf16* __restrict__ base,
                                              size_t ld, int lane, int k0) {
  const int r  = lane & 15;
  const int kg = lane >> 4;
  const __bf16* p = base + (size_t)r * ld + (size_t)(k0 + kg * 8);
  v8bf lo = *reinterpret_cast<const v8bf*>(p);
  v8bf hi = *reinterpret_cast<const v8bf*>(p + 16);
  v16bf out;
#pragma unroll
  for (int e = 0; e < 8; ++e) { out[e] = lo[e]; out[e + 8] = hi[e]; }
  return out;
}

// Same, but matrix tile lives in LDS with fixed row pitch KC.
static __device__ inline v16bf frag_from_lds(const __bf16* smem_tile,
                                             int lane, int k0rel) {
  const int col = lane & 15;
  const int kg  = lane >> 4;
  const __bf16* p = smem_tile + col * KC + k0rel + kg * 8;
  v8bf lo = *reinterpret_cast<const v8bf*>(p);
  v8bf hi = *reinterpret_cast<const v8bf*>(p + 16);
  v16bf out;
#pragma unroll
  for (int e = 0; e < 8; ++e) { out[e] = lo[e]; out[e + 8] = hi[e]; }
  return out;
}

// f32 source, converted in-register (one-shot context GEMM only).
static __device__ inline v16bf frag_from_f32(const float* __restrict__ base,
                                             size_t ld, int lane, int k0) {
  const int r  = lane & 15;
  const int kg = lane >> 4;
  const float* p = base + (size_t)r * ld + (size_t)(k0 + kg * 8);
  v16bf out;
#pragma unroll
  for (int e = 0; e < 8; ++e) {
    out[e]     = (__bf16)p[e];
    out[e + 8] = (__bf16)p[e + 16];
  }
  return out;
}

static __device__ inline float sigmoidf_(float x) {
  return 1.0f / (1.0f + __expf(-x));
}

// ---------------------------------------------------------------------------
// CDNA5 async global->LDS copy of one 16-byte quantum (ASYNCcnt-tracked).
// ---------------------------------------------------------------------------
static __device__ inline void async_copy_b128(uint32_t lds_off, uint64_t gaddr) {
  asm volatile("global_load_async_to_lds_b128 %0, %1, off"
               :: "v"(lds_off), "v"(gaddr)
               : "memory");
}
static __device__ inline void wait_asynccnt0() {
  asm volatile("s_wait_asynccnt 0x0" ::: "memory");
}

// Stage a 64-row x KC-col bf16 block of W (rows g*H+nb+r) into LDS via async
// per-lane copies. 256 threads, 4 x 16B each.
static __device__ inline void stage_B_chunk_async(const __bf16* __restrict__ W,
                                                  size_t ldw, int nb, int kc,
                                                  __bf16* smem, int tid) {
#pragma unroll
  for (int it = 0; it < 4; ++it) {
    const int idx = it * 256 + tid;   // 0..1023
    const int row = idx >> 4;         // 0..63  (gate*16 + col)
    const int seg = idx & 15;         // 16B segment (8 bf16)
    const int g   = row >> 4;
    const int r   = row & 15;
    const __bf16* gp =
        W + (size_t)(g * H_DIM + nb + r) * ldw + (size_t)(kc + seg * 8);
    const uint32_t lds_off =
        (uint32_t)(uintptr_t)(&smem[row * KC + seg * 8]);
    async_copy_b128(lds_off, (uint64_t)(uintptr_t)gp);
  }
}

// ---------------------------------------------------------------------------
// Tensor Data Mover: one 2-D tile load, 16 rows x KC bf16 cols, row stride ldw
// (elements). D# built per CDNA5 ISA. TENSORcnt-tracked.
// ---------------------------------------------------------------------------
static __device__ inline void tdm_load_16xKC(const __bf16* gbase, uint32_t ldw,
                                             uint32_t lds_off) {
  const uint64_t ga = (uint64_t)(uintptr_t)gbase;
  u32x4_t g0;
  g0[0] = 1u;                                   // count=1 (valid), user mode
  g0[1] = lds_off;                              // lds_addr (bytes)
  g0[2] = (uint32_t)ga;                         // global_addr[31:0]
  g0[3] = (uint32_t)((ga >> 32) & 0x01FFFFFFu)  // global_addr[56:32]
          | 0x80000000u;                        // type=2 ("image")
  i32x8_t g1;
  g1[0] = (int)(1u << 16);                      // wg_mask=0, data_size=1 (2B)
  g1[1] = (int)((uint32_t)KC << 16);            // tensor_dim0 = KC
  g1[2] = (int)(16u << 16);                     // tensor_dim1 = 16
  g1[3] = (int)((uint32_t)KC << 16);            // tile_dim0 = KC
  g1[4] = 16;                                   // tile_dim1 = 16, tile_dim2 = 0
  g1[5] = (int)ldw;                             // tensor_dim0_stride[31:0]
  g1[6] = 0;                                    // stride hi, dim1_stride lo
  g1[7] = 0;
  const i32x4_t z4 = {0, 0, 0, 0};
#if __clang_major__ >= 23
  const i32x8_t z8 = {0, 0, 0, 0, 0, 0, 0, 0};
  __builtin_amdgcn_tensor_load_to_lds(g0, g1, z4, z4, z8, 0);
#else
  __builtin_amdgcn_tensor_load_to_lds(g0, g1, z4, z4, 0);
#endif
}

// Stage the 4 gate tiles (16 x KC each) of W columns [nb,nb+16) at k-offset kc
// via 4 TDM descriptors. Call from a single wave.
static __device__ inline void stage_B_chunk_tdm(const __bf16* __restrict__ W,
                                                uint32_t ldw, int nb, int kc,
                                                __bf16* smem) {
#pragma unroll
  for (int g = 0; g < 4; ++g) {
    const __bf16* gp = W + (size_t)(g * H_DIM + nb) * ldw + (size_t)kc;
    const uint32_t lds_off = (uint32_t)(uintptr_t)(&smem[g * 16 * KC]);
    tdm_load_16xKC(gp, ldw, lds_off);
  }
}

// ---------------------------------------------------------------------------
// Utility kernels
// ---------------------------------------------------------------------------
__global__ void cvt_f32_bf16_kernel(const float* __restrict__ src,
                                    __bf16* __restrict__ dst, int n) {
  int i = blockIdx.x * blockDim.x + threadIdx.x;
  if (i < n) dst[i] = (__bf16)src[i];
}

// Gather+convert one timestep: x_bf[b,k] = (bf16) i_features[b, t, k]
__global__ void cvt_xt_kernel(const float* __restrict__ x_t,  // + t*I
                              __bf16* __restrict__ x_bf) {
  int i = blockIdx.x * blockDim.x + threadIdx.x;  // 0 .. B*I-1
  int row = i >> 10;        // / I_DIM
  int k   = i & (I_DIM - 1);
  x_bf[i] = (__bf16)x_t[(size_t)row * (T_DIM * I_DIM) + k];
}

__global__ void bias_sum_kernel(const float* __restrict__ bi,
                                const float* __restrict__ bh,
                                const float* __restrict__ bc,
                                float* __restrict__ bsum, int n) {
  int i = blockIdx.x * blockDim.x + threadIdx.x;
  if (i < n) bsum[i] = bi[i] + bh[i] + bc[i];
}

__global__ void init_state_kernel(__bf16* __restrict__ h0,
                                  float* __restrict__ c0, int n) {
  int i = blockIdx.x * blockDim.x + threadIdx.x;
  if (i < n) { h0[i] = (__bf16)0.0f; c0[i] = 0.0f; }
}

// ---------------------------------------------------------------------------
// base = cvt_bf16(v_features) @ Wc_bf16^T + (bi+bh+bc)   (one-shot, small)
// ---------------------------------------------------------------------------
__global__ __launch_bounds__(32) void gemm_f32a_kernel(
    const float* __restrict__ A, size_t lda, const __bf16* __restrict__ W,
    size_t ldw, const float* __restrict__ bias, float* __restrict__ out,
    int ldout, int K) {
  const int lane  = threadIdx.x;
  const int nb    = blockIdx.x * 16;
  const int mb    = blockIdx.y * 16;
  const int col   = nb + (lane & 15);
  const int mhalf = (lane >> 4) * 8;

  v8f acc;
  const float bv = bias[col];
#pragma unroll
  for (int v = 0; v < 8; ++v) acc[v] = bv;

  const float*  Ab = A + (size_t)mb * lda;
  const __bf16* Wb = W + (size_t)nb * ldw;
  for (int k0 = 0; k0 < K; k0 += 32) {
    v16bf a = frag_from_f32(Ab, lda, lane, k0);
    v16bf b = frag_from_bf16(Wb, ldw, lane, k0);
    acc = wmma_bf16(a, b, acc);
  }
#pragma unroll
  for (int v = 0; v < 8; ++v)
    out[(size_t)(mb + mhalf + v) * ldout + col] = acc[v];
}

// fc1 = relu(h_T(bf16) @ Wfa_bf16^T + bfa)
__global__ __launch_bounds__(32) void gemm_bf16a_relu_kernel(
    const __bf16* __restrict__ A, size_t lda, const __bf16* __restrict__ W,
    size_t ldw, const float* __restrict__ bias, float* __restrict__ out,
    int ldout, int K) {
  const int lane  = threadIdx.x;
  const int nb    = blockIdx.x * 16;
  const int mb    = blockIdx.y * 16;
  const int col   = nb + (lane & 15);
  const int mhalf = (lane >> 4) * 8;

  v8f acc;
  const float bv = bias[col];
#pragma unroll
  for (int v = 0; v < 8; ++v) acc[v] = bv;

  const __bf16* Ab = A + (size_t)mb * lda;
  const __bf16* Wb = W + (size_t)nb * ldw;
  for (int k0 = 0; k0 < K; k0 += 32) {
    v16bf a = frag_from_bf16(Ab, lda, lane, k0);
    v16bf b = frag_from_bf16(Wb, ldw, lane, k0);
    acc = wmma_bf16(a, b, acc);
  }
#pragma unroll
  for (int v = 0; v < 8; ++v)
    out[(size_t)(mb + mhalf + v) * ldout + col] = fmaxf(acc[v], 0.0f);
}

// ---------------------------------------------------------------------------
// Fused LSTM step.
//   grid = (64 column tiles, 2 row groups), block = 256 (8 waves).
//   Wave w owns row tile mb = blockIdx.y*128 + w*16; 8 waves share the 4-gate
//   B tiles staged in LDS, double-buffered:
//     phase 1 (x@Wi^T): per-lane async global->LDS copies  (ASYNCcnt)
//     phase 2 (h@Wh^T): Tensor Data Mover descriptors      (TENSORcnt)
// ---------------------------------------------------------------------------
__global__ __launch_bounds__(256) void lstm_step_kernel(
    const __bf16* __restrict__ x_bf,   // [B, I] bf16 (this timestep)
    const __bf16* __restrict__ Wi_bf,  // [4H, I]
    const __bf16* __restrict__ Wh_bf,  // [4H, H]
    const __bf16* __restrict__ h_in,   // [B, H] bf16
    const float* __restrict__ base,    // [B, 4H] f32 (ctx + all biases)
    float* __restrict__ c_state,       // [B, H] in/out
    __bf16* __restrict__ h_out) {      // [B, H] bf16
  __shared__ __bf16 smemB[2][64 * KC];

  const int tid   = threadIdx.x;
  const int lane  = tid & 31;
  const int wave  = tid >> 5;                 // 0..7
  const int nb    = blockIdx.x * 16;          // column tile within H
  const int mb    = blockIdx.y * 128 + wave * 16;
  const int col   = nb + (lane & 15);
  const int mhalf = (lane >> 4) * 8;

  // Seed accumulators with base (= ctx_proj + bi + bh + bc), gate-chunked.
  v8f acc[4];
#pragma unroll
  for (int g = 0; g < 4; ++g)
#pragma unroll
    for (int v = 0; v < 8; ++v)
      acc[g][v] = base[(size_t)(mb + mhalf + v) * G4H + g * H_DIM + col];

  int p = 0;
  // -------- phase 1: x @ Wi^T  (async global->LDS staging) --------
  stage_B_chunk_async(Wi_bf, I_DIM, nb, 0, smemB[p], tid);
  const __bf16* arow_x = x_bf + (size_t)mb * I_DIM;
  for (int kc = 0; kc < I_DIM; kc += KC) {
    wait_asynccnt0();
    __syncthreads();
    if (kc + KC < I_DIM)
      stage_B_chunk_async(Wi_bf, I_DIM, nb, kc + KC, smemB[p ^ 1], tid);
#pragma unroll
    for (int k0 = 0; k0 < KC; k0 += 32) {
      v16bf a = frag_from_bf16(arow_x, I_DIM, lane, kc + k0);
#pragma unroll
      for (int g = 0; g < 4; ++g) {
        v16bf b = frag_from_lds(&smemB[p][g * 16 * KC], lane, k0);
        acc[g] = wmma_bf16(a, b, acc[g]);
      }
    }
    __syncthreads();
    p ^= 1;
  }
  // -------- phase 2: h @ Wh^T  (TDM staging, wave 0 drives the DMA) --------
  if (wave == 0) stage_B_chunk_tdm(Wh_bf, H_DIM, nb, 0, smemB[p]);
  const __bf16* arow_h = h_in + (size_t)mb * H_DIM;
  for (int kc = 0; kc < H_DIM; kc += KC) {
    if (wave == 0) __builtin_amdgcn_s_wait_tensorcnt(0);
    __syncthreads();
    if (kc + KC < H_DIM && wave == 0)
      stage_B_chunk_tdm(Wh_bf, H_DIM, nb, kc + KC, smemB[p ^ 1]);
#pragma unroll
    for (int k0 = 0; k0 < KC; k0 += 32) {
      v16bf a = frag_from_bf16(arow_h, H_DIM, lane, kc + k0);
#pragma unroll
      for (int g = 0; g < 4; ++g) {
        v16bf b = frag_from_lds(&smemB[p][g * 16 * KC], lane, k0);
        acc[g] = wmma_bf16(a, b, acc[g]);
      }
    }
    __syncthreads();
    p ^= 1;
  }

  // Pointwise LSTM update; C/D layout: N = lane&15, M = v + 8*(lane>>4).
#pragma unroll
  for (int v = 0; v < 8; ++v) {
    const int    m   = mb + mhalf + v;
    const size_t idx = (size_t)m * H_DIM + col;
    const float ig = sigmoidf_(acc[0][v]);
    const float fg = sigmoidf_(acc[1][v]);
    const float cg = tanhf(acc[2][v]);
    const float og = sigmoidf_(acc[3][v]);
    const float cn = fg * c_state[idx] + ig * cg;
    c_state[idx] = cn;
    h_out[idx]   = (__bf16)(og * tanhf(cn));
  }
}

// ---------------------------------------------------------------------------
// Tiny classifier head: out[b,c] = fc1[b,:] . Wfc[c,:] + bfc[c]
// ---------------------------------------------------------------------------
__global__ void fc2_kernel(const float* __restrict__ x,
                           const float* __restrict__ Wfc,
                           const float* __restrict__ bfc,
                           float* __restrict__ out) {
  int idx = blockIdx.x * blockDim.x + threadIdx.x;
  if (idx >= B_DIM * NC_DIM) return;
  int b = idx / NC_DIM, c = idx % NC_DIM;
  const float* xr = x + (size_t)b * F_DIM;
  const float* wr = Wfc + (size_t)c * F_DIM;
  float s = bfc[c];
  for (int k = 0; k < F_DIM; ++k) s += xr[k] * wr[k];
  out[idx] = s;
}

// ---------------------------------------------------------------------------
extern "C" void kernel_launch(void* const* d_in, const int* in_sizes, int n_in,
                              void* d_out, int out_size, void* d_ws,
                              size_t ws_size, hipStream_t stream) {
  const float* v_feat = (const float*)d_in[0];
  const float* i_feat = (const float*)d_in[1];
  const float* Wi  = (const float*)d_in[2];
  const float* bi  = (const float*)d_in[3];
  const float* Wh  = (const float*)d_in[4];
  const float* bh  = (const float*)d_in[5];
  const float* Wc  = (const float*)d_in[6];
  const float* bc  = (const float*)d_in[7];
  const float* Wfa = (const float*)d_in[8];
  const float* bfa = (const float*)d_in[9];
  const float* Wfc = (const float*)d_in[10];
  const float* bfc = (const float*)d_in[11];
  float* out = (float*)d_out;

  // Workspace carving (~35 MB total)
  char* ws = (char*)d_ws;
  auto carve = [&](size_t bytes) -> char* {
    char* p = ws;
    ws += (bytes + 255) & ~(size_t)255;
    return p;
  };
  __bf16* Wi_bf  = (__bf16*)carve(sizeof(__bf16) * (size_t)G4H * I_DIM);
  __bf16* Wh_bf  = (__bf16*)carve(sizeof(__bf16) * (size_t)G4H * H_DIM);
  __bf16* Wc_bf  = (__bf16*)carve(sizeof(__bf16) * (size_t)G4H * C_DIM);
  __bf16* Wfa_bf = (__bf16*)carve(sizeof(__bf16) * (size_t)F_DIM * H_DIM);
  float*  bsum   = (float*)carve(sizeof(float) * G4H);
  float*  base   = (float*)carve(sizeof(float) * (size_t)B_DIM * G4H);
  __bf16* x_bf   = (__bf16*)carve(sizeof(__bf16) * (size_t)B_DIM * I_DIM);
  __bf16* h_bf0  = (__bf16*)carve(sizeof(__bf16) * (size_t)B_DIM * H_DIM);
  __bf16* h_bf1  = (__bf16*)carve(sizeof(__bf16) * (size_t)B_DIM * H_DIM);
  float*  c_buf  = (float*)carve(sizeof(float) * (size_t)B_DIM * H_DIM);
  float*  fc1    = (float*)carve(sizeof(float) * (size_t)B_DIM * F_DIM);

  const int TPB = 256;
  // 1) weight conversions fp32 -> bf16
  {
    int n = G4H * I_DIM;
    cvt_f32_bf16_kernel<<<(n + TPB - 1) / TPB, TPB, 0, stream>>>(Wi, Wi_bf, n);
    n = G4H * H_DIM;
    cvt_f32_bf16_kernel<<<(n + TPB - 1) / TPB, TPB, 0, stream>>>(Wh, Wh_bf, n);
    n = G4H * C_DIM;
    cvt_f32_bf16_kernel<<<(n + TPB - 1) / TPB, TPB, 0, stream>>>(Wc, Wc_bf, n);
    n = F_DIM * H_DIM;
    cvt_f32_bf16_kernel<<<(n + TPB - 1) / TPB, TPB, 0, stream>>>(Wfa, Wfa_bf, n);
  }
  // 2) combined bias
  bias_sum_kernel<<<(G4H + TPB - 1) / TPB, TPB, 0, stream>>>(bi, bh, bc, bsum,
                                                             G4H);
  // 3) base = v_features @ Wc.T + (bi+bh+bc)   [B, 4H]
  gemm_f32a_kernel<<<dim3(G4H / 16, B_DIM / 16), 32, 0, stream>>>(
      v_feat, (size_t)C_DIM, Wc_bf, (size_t)C_DIM, bsum, base, G4H, C_DIM);
  // 4) zero-init h, c
  {
    int n = B_DIM * H_DIM;
    init_state_kernel<<<(n + TPB - 1) / TPB, TPB, 0, stream>>>(h_bf0, c_buf, n);
  }
  // 5) recurrence: per step convert x_t then fused step; ping-pong h buffers
  __bf16* h_in  = h_bf0;
  __bf16* h_out = h_bf1;
  for (int t = 0; t < T_DIM; ++t) {
    const float* x_t = i_feat + (size_t)t * I_DIM;  // row stride = T*I
    cvt_xt_kernel<<<(B_DIM * I_DIM) / TPB, TPB, 0, stream>>>(x_t, x_bf);
    lstm_step_kernel<<<dim3(H_DIM / 16, B_DIM / 128), 256, 0, stream>>>(
        x_bf, Wi_bf, Wh_bf, h_in, base, c_buf, h_out);
    __bf16* tmp = h_in; h_in = h_out; h_out = tmp;
  }
  // h_in now holds h_T (bf16)
  // 6) fc1 = relu(h_T @ Wfa.T + bfa)   [B, F]
  gemm_bf16a_relu_kernel<<<dim3(F_DIM / 16, B_DIM / 16), 32, 0, stream>>>(
      h_in, (size_t)H_DIM, Wfa_bf, (size_t)H_DIM, bfa, fc1, F_DIM, H_DIM);
  // 7) out = fc1 @ Wfc.T + bfc        [B, NC]
  {
    int n = B_DIM * NC_DIM;
    fc2_kernel<<<(n + 63) / 64, 64, 0, stream>>>(fc1, Wfc, bfc, out);
  }
}
